// PerspectronEncoderNetwork_43078521979619
// MI455X (gfx1250) — matrix operand order repaired and checked by reading, not compiled, not run on hardware
//
#include <hip/hip_runtime.h>
#include <hip/hip_bf16.h>

// ---------------------------------------------------------------------------
// Perspectron encoder network on MI455X (gfx1250, wave32, WMMA f16 16x16x32)
//   64 batches x (S=1024, D=256 -> L=512) : QKV gemms + 2 flash-attention
//   layers + mean-pool head + log_softmax(K=10).
// ---------------------------------------------------------------------------

typedef __attribute__((ext_vector_type(16))) _Float16 v16h;
typedef __attribute__((ext_vector_type(8)))  float    v8f;

union FragU { v16h h; unsigned int u[8]; };

__device__ __forceinline__ v8f wmma_f16(v16h a, v16h b, v8f c) {
  return __builtin_amdgcn_wmma_f32_16x16x32_f16(
      /*neg_a=*/false, a, /*neg_b=*/false, b,
      /*c_mod=*/(short)0, c, /*reuse_a=*/false, /*reuse_b=*/false);
}

// A fragment: 16(M) x 32(K) f16 tile, row-major source, ld f16-elements/row.
// lane&15 = M row; lane>>4 -> K half (+16); one contiguous 16-half chunk.
__device__ __forceinline__ v16h load_afrag(const _Float16* __restrict__ p,
                                           int ld, int lane) {
  int row = lane & 15;
  int kb  = (lane >> 4) << 4;  // 0 or 16
  const _Float16* r0 = p + (size_t)row * ld + kb;
  FragU f;
#pragma unroll
  for (int i = 0; i < 4; ++i) {
    f.u[i]     = *(const unsigned int*)(r0 + 2 * i);
    f.u[4 + i] = *(const unsigned int*)(r0 + 8 + 2 * i);
  }
  return f.h;
}

// B fragment: 32(K) x 16(N). Source is B^T stored row-major [n][k]
// (k contiguous). lane&15 = N col; lane>>4 -> K half (+16);
// one contiguous 16-half chunk per lane.
__device__ __forceinline__ v16h load_bfrag(const _Float16* __restrict__ p,
                                           int ld, int lane) {
  int n  = lane & 15;
  int kh = (lane >> 4) << 4;   // 0 or 16
  const _Float16* r0 = p + (size_t)n * ld + kh;
  FragU f;
#pragma unroll
  for (int i = 0; i < 8; ++i) f.u[i] = *(const unsigned int*)(r0 + 2 * i);
  return f.h;
}

// C/D accumulator element at (lane, reg r): M = r + 8*(lane>>4), N = lane&15.

// ---------------------------------------------------------------------------
// Prep kernels
// ---------------------------------------------------------------------------

__global__ __launch_bounds__(256) void concat_convert_x(
    const float* __restrict__ x, const float* __restrict__ y,
    _Float16* __restrict__ xf) {
  size_t i = (size_t)blockIdx.x * 256 + threadIdx.x;  // over 65536*256
  size_t m = i >> 8;
  int    c = (int)(i & 255);
  float v = (c < 255) ? x[m * 255 + c] : y[m];
  xf[i] = (_Float16)v;
}

// W [K][N] f32 -> W^T f16 [N][K]
__global__ __launch_bounds__(256) void transpose_convert_w(
    const float* __restrict__ W, _Float16* __restrict__ Wt, int K, int N) {
  int i = blockIdx.x * 256 + threadIdx.x;
  if (i >= K * N) return;
  int k = i / N, n = i % N;
  Wt[(size_t)n * K + k] = (_Float16)W[i];
}

__global__ __launch_bounds__(256) void zero_f32(float* __restrict__ p, int n) {
  int i = blockIdx.x * 256 + threadIdx.x;
  if (i < n) p[i] = 0.f;
}

__global__ __launch_bounds__(256) void phi_finalize(
    const float* __restrict__ phiSum, _Float16* __restrict__ phif) {
  int i = blockIdx.x * 256 + threadIdx.x;  // 64*512
  float v = phiSum[i] * (1.f / 1024.f);    // mean over S
  phif[i] = (_Float16)fmaxf(v, 0.f);       // relu AFTER mean
}

__global__ __launch_bounds__(256) void logsoftmax_k(
    const float* __restrict__ logits, float* __restrict__ out) {
  int i = blockIdx.x * 256 + threadIdx.x;  // 64*512 groups of K=10
  if (i >= 64 * 512) return;
  const float* g = logits + (size_t)i * 10;
  float mx = g[0];
#pragma unroll
  for (int k = 1; k < 10; ++k) mx = fmaxf(mx, g[k]);
  float s = 0.f;
#pragma unroll
  for (int k = 0; k < 10; ++k) s += __expf(g[k] - mx);
  float lse = mx + __logf(s);
#pragma unroll
  for (int k = 0; k < 10; ++k) out[(size_t)i * 10 + k] = g[k] - lse;
}

// ---------------------------------------------------------------------------
// Generic WMMA GEMM:  C[M][N] = A[M][K](f16) x Bt[N][K](f16)^T + bias
// Block tile 64x256, 8 waves in 2(M) x 4(N); each wave owns a 32x64 tile
// (2x4 accumulators, 8 v_wmma per 32-deep k-step, B frags reused across M).
//   OutF != null : store f32 row-major [M][N]
//   transToVt    : store f16 transposed per batch  [b][col(512)][s(1024)]
//   else         : store f16 row-major [M][N]
// ---------------------------------------------------------------------------
__global__ __launch_bounds__(256) void gemm_f16(
    const _Float16* __restrict__ A, const _Float16* __restrict__ Bt,
    const float* __restrict__ bias, _Float16* __restrict__ OutH,
    float* __restrict__ OutF, int N, int K, int transToVt) {
  int m0 = blockIdx.x * 64;
  int n0 = blockIdx.y * 256;
  int tid = threadIdx.x, lane = tid & 31, w = tid >> 5;
  int wm = w & 1, wn = w >> 1;
  const _Float16* Arow0 = A + (size_t)(m0 + wm * 32) * K;
  const _Float16* Brow0 = Bt + (size_t)(n0 + wn * 64) * K;
  v8f acc[2][4] = {};
  for (int k0 = 0; k0 < K; k0 += 32) {
    __builtin_prefetch(Arow0 + k0 + 128, 0, 1);
    __builtin_prefetch(Brow0 + k0 + 128, 0, 1);
    v16h a0 = load_afrag(Arow0 + k0, K, lane);
    v16h a1 = load_afrag(Arow0 + (size_t)16 * K + k0, K, lane);
#pragma unroll
    for (int j = 0; j < 4; ++j) {
      v16h bj = load_bfrag(Brow0 + (size_t)(16 * j) * K + k0, K, lane);
      acc[0][j] = wmma_f16(a0, bj, acc[0][j]);
      acc[1][j] = wmma_f16(a1, bj, acc[1][j]);
    }
  }
  int lh = lane >> 4, cn = lane & 15;
#pragma unroll
  for (int i = 0; i < 2; ++i) {
#pragma unroll
    for (int j = 0; j < 4; ++j) {
      int col = n0 + wn * 64 + j * 16 + cn;
      float bv = bias ? bias[col] : 0.f;
#pragma unroll
      for (int r = 0; r < 8; ++r) {
        int row = m0 + wm * 32 + i * 16 + r + 8 * lh;
        float v = acc[i][j][r] + bv;
        if (OutF) {
          OutF[(size_t)row * N + col] = v;
        } else if (transToVt) {
          int bb = row >> 10, sl = row & 1023;      // row = b*1024 + s
          OutH[((size_t)bb * 512 + col) * 1024 + sl] = (_Float16)v;
        } else {
          OutH[(size_t)row * N + col] = (_Float16)v;
        }
      }
    }
  }
}

// ---------------------------------------------------------------------------
// Flash attention: per block 32 queries of one batch, full 1024-key sweep in
// 64-key tiles.  All 8 waves (2x4) each compute one 16x16 tile of the 32x64
// score block (16 wmma each), wave 0 does the online softmax, then all 8
// waves (2(M) x 4(Dcols)) rescale and accumulate O += P @ V (16 wmma each).
//   pool == 0 : h = relu(O / l)      -> Hout f16 [b*S+q][512]
//   pool == 1 : column-sum of O / l  -> atomicAdd into phiSum [b][512]
// ---------------------------------------------------------------------------
#define ATT_S 1024
#define ATT_D 512
#define KV_T  64

__global__ __launch_bounds__(256) void attn_kernel(
    const _Float16* __restrict__ Q, const _Float16* __restrict__ Km,
    const _Float16* __restrict__ Vt, _Float16* __restrict__ Hout,
    float* __restrict__ phiSum, int pool) {
  __shared__ float    Ss[32][65];     // 32x64 score tile (padded)
  __shared__ _Float16 Ps[32][72];     // softmaxed probs f16 (padded ld)
  __shared__ float    mrow[32], lrow[32], rowscale[32];

  int tid = threadIdx.x;
  int lane = tid & 31;
  int w = tid >> 5;          // 0..7
  int mi = w & 1;            // query half (16 rows)
  int cg = w >> 1;           // 0..3 : 128-wide O-column group / 16-key S col
  int blk = blockIdx.x;      // 64 batches * 32 query tiles
  int b = blk >> 5;
  int q0 = (blk & 31) << 5;

  if (tid < 32) { mrow[tid] = -3.0e38f; lrow[tid] = 0.f; }
  __syncthreads();

  v8f acc[8] = {};  // 16 x 128 f32 output accumulator per wave

  const _Float16* Qb = Q  + (size_t)(b * ATT_S + q0) * ATT_D;
  const _Float16* Kb = Km + (size_t)b * ATT_S * ATT_D;
  const _Float16* Vb = Vt + (size_t)b * ATT_D * ATT_S;  // [d][s]
  const float rs = 0.04419417382415922f;                // 1/sqrt(512)

  for (int kv = 0; kv < ATT_S / KV_T; ++kv) {
    // ---- scores: wave (mi,cg) computes S[mi*16..][cg*16..] over k=512 -----
    {
      v8f sacc = {};
      const _Float16* qa = Qb + (size_t)(mi * 16) * ATT_D;
      const _Float16* kb = Kb + (size_t)(kv * KV_T + cg * 16) * ATT_D;
#pragma unroll 4
      for (int kt = 0; kt < ATT_D / 32; ++kt) {
        __builtin_prefetch(kb + kt * 32 + 128, 0, 1);
        v16h a  = load_afrag(qa + kt * 32, ATT_D, lane);
        v16h bb = load_bfrag(kb + kt * 32, ATT_D, lane);
        sacc = wmma_f16(a, bb, sacc);
      }
      int cn = lane & 15, lh = lane >> 4;
#pragma unroll
      for (int r = 0; r < 8; ++r)
        Ss[mi * 16 + r + 8 * lh][cg * 16 + cn] = sacc[r] * rs;
    }
    __syncthreads();

    // ---- online softmax: thread t owns query row t ------------------------
    if (tid < 32) {
      float mOld = mrow[tid];
      float mx = mOld;
#pragma unroll 8
      for (int j = 0; j < KV_T; ++j) mx = fmaxf(mx, Ss[tid][j]);
      float sc = __expf(mOld - mx);
      float l = lrow[tid] * sc;
#pragma unroll 8
      for (int j = 0; j < KV_T; ++j) {
        float pv = __expf(Ss[tid][j] - mx);
        l += pv;
        Ps[tid][j] = (_Float16)pv;
      }
      mrow[tid] = mx; lrow[tid] = l; rowscale[tid] = sc;
    }
    __syncthreads();

    // ---- rescale O, O += P @ V  (k = 64 keys -> 2 wmma per n-tile) --------
    {
      int lh = lane >> 4;
#pragma unroll
      for (int j = 0; j < 8; ++j) {
        v8f t = acc[j];
#pragma unroll
        for (int r = 0; r < 8; ++r) t[r] *= rowscale[mi * 16 + r + 8 * lh];
        acc[j] = t;
      }
      v16h pA0 = load_afrag(&Ps[mi * 16][0], 72, lane);
      v16h pA1 = load_afrag(&Ps[mi * 16][32], 72, lane);
      const _Float16* vb0 = Vb + (size_t)(cg * 128) * ATT_S + kv * KV_T;
#pragma unroll
      for (int j = 0; j < 8; ++j) {
        const _Float16* vr = vb0 + (size_t)(j * 16) * ATT_S;
        v16h v0 = load_bfrag(vr, ATT_S, lane);
        v16h v1 = load_bfrag(vr + 32, ATT_S, lane);
        acc[j] = wmma_f16(pA0, v0, acc[j]);
        acc[j] = wmma_f16(pA1, v1, acc[j]);
      }
    }
    __syncthreads();  // protect Ps/rowscale before next tile's softmax
  }

  // ---- epilogue -----------------------------------------------------------
  int lh = lane >> 4, cn = lane & 15;
  if (!pool) {
#pragma unroll
    for (int j = 0; j < 8; ++j) {
      int col = cg * 128 + j * 16 + cn;
#pragma unroll
      for (int r = 0; r < 8; ++r) {
        int row = mi * 16 + r + 8 * lh;
        float v = acc[j][r] / lrow[row];
        Hout[(size_t)(b * ATT_S + q0 + row) * ATT_D + col] =
            (_Float16)fmaxf(v, 0.f);  // relu for layer-1 h
      }
    }
  } else {
#pragma unroll
    for (int j = 0; j < 8; ++j) {
      float cs = 0.f;
#pragma unroll
      for (int r = 0; r < 8; ++r) {
        int row = mi * 16 + r + 8 * lh;
        cs += acc[j][r] / lrow[row];
      }
      cs += __shfl_xor(cs, 16, 32);  // combine the two lane halves (same col)
      if (lane < 16)
        atomicAdd(&phiSum[b * ATT_D + cg * 128 + j * 16 + cn], cs);
    }
  }
}

// ---------------------------------------------------------------------------
// Host launch
// ---------------------------------------------------------------------------
extern "C" void kernel_launch(void* const* d_in, const int* in_sizes, int n_in,
                              void* d_out, int out_size, void* d_ws,
                              size_t ws_size, hipStream_t stream) {
  (void)in_sizes; (void)n_in; (void)out_size; (void)ws_size;
  const float* x_sets = (const float*)d_in[0];
  const float* y_sets = (const float*)d_in[1];
  const float* Wq1 = (const float*)d_in[2];  const float* bq1 = (const float*)d_in[3];
  const float* Wk1 = (const float*)d_in[4];  const float* bk1 = (const float*)d_in[5];
  const float* Wv1 = (const float*)d_in[6];  const float* bv1 = (const float*)d_in[7];
  const float* Wq2 = (const float*)d_in[8];  const float* bq2 = (const float*)d_in[9];
  const float* Wk2 = (const float*)d_in[10]; const float* bk2 = (const float*)d_in[11];
  const float* Wv2 = (const float*)d_in[12]; const float* bv2 = (const float*)d_in[13];
  const float* We  = (const float*)d_in[14]; const float* be  = (const float*)d_in[15];
  float* out = (float*)d_out;

  char* ws = (char*)d_ws;
  size_t off = 0;
  auto take = [&](size_t bytes) -> char* {
    char* p = ws + off;
    off += (bytes + 255) & ~(size_t)255;
    return p;
  };
  _Float16* xf    = (_Float16*)take(65536ull * 256 * 2);  // concat(x,y) f16
  _Float16* wtq1  = (_Float16*)take(512ull * 256 * 2);
  _Float16* wtk1  = (_Float16*)take(512ull * 256 * 2);
  _Float16* wtv1  = (_Float16*)take(512ull * 256 * 2);
  _Float16* wtq2  = (_Float16*)take(512ull * 512 * 2);
  _Float16* wtk2  = (_Float16*)take(512ull * 512 * 2);
  _Float16* wtv2  = (_Float16*)take(512ull * 512 * 2);
  _Float16* wte   = (_Float16*)take(5120ull * 512 * 2);
  _Float16* qf    = (_Float16*)take(65536ull * 512 * 2);  // Q  [b*s][512]
  _Float16* kf    = (_Float16*)take(65536ull * 512 * 2);  // K  [b*s][512]
  _Float16* vtf   = (_Float16*)take(65536ull * 512 * 2);  // V^T [b][512][1024]
  _Float16* hf    = (_Float16*)take(65536ull * 512 * 2);  // h  [b*s][512]
  float*    phiS  = (float*)   take(64ull * 512 * 4);
  _Float16* phif  = (_Float16*)take(64ull * 512 * 2);
  float*    logit = (float*)   take(64ull * 5120 * 4);

  // ---- prep ----------------------------------------------------------------
  concat_convert_x<<<65536, 256, 0, stream>>>(x_sets, y_sets, xf);
  transpose_convert_w<<<(256 * 512 + 255) / 256, 256, 0, stream>>>(Wq1, wtq1, 256, 512);
  transpose_convert_w<<<(256 * 512 + 255) / 256, 256, 0, stream>>>(Wk1, wtk1, 256, 512);
  transpose_convert_w<<<(256 * 512 + 255) / 256, 256, 0, stream>>>(Wv1, wtv1, 256, 512);
  transpose_convert_w<<<(512 * 512 + 255) / 256, 256, 0, stream>>>(Wq2, wtq2, 512, 512);
  transpose_convert_w<<<(512 * 512 + 255) / 256, 256, 0, stream>>>(Wk2, wtk2, 512, 512);
  transpose_convert_w<<<(512 * 512 + 255) / 256, 256, 0, stream>>>(Wv2, wtv2, 512, 512);
  transpose_convert_w<<<(512 * 5120 + 255) / 256, 256, 0, stream>>>(We, wte, 512, 5120);

  // ---- layer 1 -------------------------------------------------------------
  dim3 gQKV(65536 / 64, 512 / 256);  // (1024, 2)
  gemm_f16<<<gQKV, 256, 0, stream>>>(xf, wtq1, bq1, qf,  nullptr, 512, 256, 0);
  gemm_f16<<<gQKV, 256, 0, stream>>>(xf, wtk1, bk1, kf,  nullptr, 512, 256, 0);
  gemm_f16<<<gQKV, 256, 0, stream>>>(xf, wtv1, bv1, vtf, nullptr, 512, 256, 1);
  attn_kernel<<<2048, 256, 0, stream>>>(qf, kf, vtf, hf, nullptr, 0);

  // ---- layer 2 -------------------------------------------------------------
  gemm_f16<<<gQKV, 256, 0, stream>>>(hf, wtq2, bq2, qf,  nullptr, 512, 512, 0);
  gemm_f16<<<gQKV, 256, 0, stream>>>(hf, wtk2, bk2, kf,  nullptr, 512, 512, 0);
  gemm_f16<<<gQKV, 256, 0, stream>>>(hf, wtv2, bv2, vtf, nullptr, 512, 512, 1);
  zero_f32<<<(64 * 512) / 256, 256, 0, stream>>>(phiS, 64 * 512);
  attn_kernel<<<2048, 256, 0, stream>>>(qf, kf, vtf, nullptr, phiS, 1);
  phi_finalize<<<(64 * 512) / 256, 256, 0, stream>>>(phiS, phif);

  // ---- head ----------------------------------------------------------------
  dim3 gHead(1, 5120 / 256);  // M=64 fits one 64-row block
  gemm_f16<<<gHead, 256, 0, stream>>>(phif, wte, be, nullptr, logit, 5120, 512, 0);
  logsoftmax_k<<<(64 * 512 + 255) / 256, 256, 0, stream>>>(logit, out);
}
